// Block_74217034875661
// MI455X (gfx1250) — matrix operand order repaired and checked by reading, not compiled, not run on hardware
//
#include <hip/hip_runtime.h>
#include <math.h>

// ---------------- problem constants ----------------
#define Bc   2
#define Sc   1024
#define Dc   2048
#define Hc   16
#define HDc  128
#define ROTc 64
#define LQc  1024
#define LKVc 512
#define EDc  1024
#define Ec   8
#define Tc   (Bc*Sc)        // 2048 tokens
#define DHc  (Hc*HDc)       // 2048
#define CATc (HDc+ROTc)     // 192

// ---------------- WMMA types ----------------
typedef __attribute__((ext_vector_type(16))) __bf16 v16bf;
typedef __attribute__((ext_vector_type(8)))  float  v8f;

struct __attribute__((aligned(16))) U16x8 { unsigned int a, b, c, d; };
struct Q2 { U16x8 lo, hi; };

__device__ __forceinline__ unsigned short f2bf(float f) {
  unsigned int u = __float_as_uint(f);
  u += 0x7FFFu + ((u >> 16) & 1u);   // round-to-nearest-even
  return (unsigned short)(u >> 16);
}
__device__ __forceinline__ float geluf(float v) {
  return 0.5f * v * (1.0f + erff(v * 0.70710678118654752440f));
}
__device__ __forceinline__ v16bf mkfrag(const unsigned short* p0, const unsigned short* p1) {
  Q2 q;
  q.lo = *(const U16x8*)p0;
  q.hi = *(const U16x8*)p1;
  return __builtin_bit_cast(v16bf, q);
}

// ---------------- CDNA5 async global->LDS copy (16B per lane) ----------------
#if __has_builtin(__builtin_amdgcn_global_load_async_to_lds_b128) && \
    __has_builtin(__builtin_amdgcn_s_wait_asynccnt)
#define USE_ASYNC 1
#else
#define USE_ASYNC 0
#endif

#define AS3 __attribute__((address_space(3)))
#define AS1 __attribute__((address_space(1)))
typedef int v4i_vs __attribute__((vector_size(16)));   // matches builtin param pointee

__device__ __forceinline__ void cp16(const unsigned short* g, unsigned short* l) {
#if USE_ASYNC
  // low 32 bits of an LDS flat address are the LDS byte offset (ISA 10.2 aperture rule)
  __builtin_amdgcn_global_load_async_to_lds_b128(
      (AS1 v4i_vs*)(uintptr_t)g, (AS3 v4i_vs*)(uintptr_t)l, 0, 0);
#else
  *(U16x8*)l = *(const U16x8*)g;
#endif
}
__device__ __forceinline__ void cp_wait() {
#if USE_ASYNC
  __builtin_amdgcn_s_wait_asynccnt(0);
#endif
}

// ---------------- generic bf16 WMMA GEMM ----------------
// C[M,N] = act( alpha * A[M,K] @ Bt[N,K]^T + bias + addsrc )
// B is supplied TRANSPOSED (row n holds K contiguous values) so both tiles
// stage into LDS with pure async b128 copies (no in-kernel transpose).
// Double-buffered LDS: tile k+1 streams in (ASYNCcnt) while tile k computes.
// A rows optionally gathered via rowIdx; M optionally bounded by *cntPtr.
// Output: Cf (f32) and/or Cb (bf16), or scatter-add into sout with gate.
#define BM 128
#define BN 128
#define BK 32

__global__ __launch_bounds__(256) void gemm_bf16(
    const unsigned short* __restrict__ A, int lda,
    const unsigned short* __restrict__ Bt, int ldbt,
    int M, int N, int K,
    const int* __restrict__ rowIdx, const int* __restrict__ cntPtr,
    const float* __restrict__ bias, const float* __restrict__ addsrc,
    float alpha, int act,
    float* __restrict__ Cf, unsigned short* __restrict__ Cb, int ldc,
    const int* __restrict__ stok, const float* __restrict__ sgate,
    float* __restrict__ sout)
{
  __shared__ __align__(16) unsigned short lA[2][BM * BK];
  __shared__ __align__(16) unsigned short lB[2][BN * BK];

  int Mb = cntPtr ? *cntPtr : M;
  int rowBase = blockIdx.y * BM;
  if (rowBase >= Mb) return;
  int nBase = blockIdx.x * BN;
  int t = threadIdx.x;

  // A tile loader: 128 rows x 32 cols, two b128 per thread
  int arow  = t >> 1;
  int acolc = (t & 1) << 4;                        // 0 or 16
  int gr = rowBase + arow;
  if (gr >= Mb) gr = Mb - 1;                       // clamp (store masked later)
  int srcRow = rowIdx ? rowIdx[gr] : gr;
  const unsigned short* aptr = A + (size_t)srcRow * lda + acolc;

  // B tile loader: 128 n-rows x 32 k, two b128 per thread
  int bn  = t >> 1;
  int bkc = (t & 1) << 4;                          // 0 or 16
  const unsigned short* bptr = Bt + (size_t)(nBase + bn) * ldbt + bkc;

  int lane = t & 31, wave = t >> 5;
  int wmR = wave >> 1, wnC = wave & 1;             // 4 x 2 wave grid, wave tile 32x64
  int lg = lane >> 4, lr = lane & 15;

  v8f acc[2][4];
#pragma unroll
  for (int i = 0; i < 2; ++i)
#pragma unroll
    for (int j = 0; j < 4; ++j)
#pragma unroll
      for (int r = 0; r < 8; ++r) acc[i][j][r] = 0.0f;

  auto issue = [&](int k0, int buf) {
    cp16(aptr + k0,     &lA[buf][arow * BK + acolc]);
    cp16(aptr + k0 + 8, &lA[buf][arow * BK + acolc + 8]);
    cp16(bptr + k0,     &lB[buf][bn * BK + bkc]);
    cp16(bptr + k0 + 8, &lB[buf][bn * BK + bkc + 8]);
  };

  int nk = K / BK;
  issue(0, 0);
  for (int kk = 0; kk < nk; ++kk) {
    int cur = kk & 1;
    cp_wait();               // tile kk resident in LDS (this wave's copies)
    __syncthreads();         // ...and everyone else's; also old buffer fully consumed
    if (kk + 1 < nk) issue((kk + 1) * BK, cur ^ 1);  // prefetch next tile (other buffer)

    v16bf af[2], bfv[4];
#pragma unroll
    for (int i = 0; i < 2; ++i) {
      int m = wmR * 32 + i * 16 + lr;
      af[i] = mkfrag(&lA[cur][m * BK + lg * 8], &lA[cur][m * BK + 16 + lg * 8]);
    }
#pragma unroll
    for (int j = 0; j < 4; ++j) {
      int n = wnC * 64 + j * 16 + lr;
      bfv[j] = mkfrag(&lB[cur][n * BK + lg * 8], &lB[cur][n * BK + 16 + lg * 8]);
    }
#pragma unroll
    for (int i = 0; i < 2; ++i)
#pragma unroll
      for (int j = 0; j < 4; ++j)
        acc[i][j] = __builtin_amdgcn_wmma_f32_16x16x32_bf16(
            false, af[i], false, bfv[j], (short)0, acc[i][j], false, false);
  }

#pragma unroll
  for (int i = 0; i < 2; ++i) {
#pragma unroll
    for (int j = 0; j < 4; ++j) {
#pragma unroll
      for (int r = 0; r < 8; ++r) {
        int row = rowBase + wmR * 32 + i * 16 + lg * 8 + r;
        int col = nBase + wnC * 64 + j * 16 + lr;
        if (row < Mb) {
          float v = acc[i][j][r] * alpha;
          if (bias)   v += bias[col];
          if (addsrc) v += addsrc[(size_t)row * ldc + col];
          if (act == 1) v = geluf(v);
          if (sout) {
            int tok = stok[row];
            atomicAdd(&sout[(size_t)tok * ldc + col], v * sgate[row]);
          } else {
            if (Cf) Cf[(size_t)row * ldc + col] = v;
            if (Cb) Cb[(size_t)row * ldc + col] = f2bf(v);
          }
        }
      }
    }
  }
}

// ---------------- elementwise / reduction kernels ----------------
__global__ void cvt_bf16k(const float* __restrict__ in, unsigned short* __restrict__ out, int n) {
  int i = blockIdx.x * 256 + threadIdx.x;
  if (i < n) out[i] = f2bf(in[i]);
}

// weight [K,N] (row-major) -> bf16 transposed [N,K]
__global__ void cvt_bf16_t(const float* __restrict__ in, unsigned short* __restrict__ out,
                           int K, int N) {
  int idx = blockIdx.x * 256 + threadIdx.x;
  if (idx >= K * N) return;
  int k = idx % K;
  int n = idx / K;
  out[idx] = f2bf(in[(size_t)k * N + n]);
}

__global__ __launch_bounds__(256) void rmsnorm_k(const float* __restrict__ x, const float* __restrict__ w,
                                                 unsigned short* __restrict__ ob, float* __restrict__ of) {
  __shared__ float red[256];
  int row = blockIdx.x;
  const float* xr = x + (size_t)row * Dc;
  int base = threadIdx.x * 8;
  float v[8], ss = 0.0f;
#pragma unroll
  for (int j = 0; j < 8; ++j) { v[j] = xr[base + j]; ss += v[j] * v[j]; }
  red[threadIdx.x] = ss; __syncthreads();
  for (int s = 128; s > 0; s >>= 1) { if ((int)threadIdx.x < s) red[threadIdx.x] += red[threadIdx.x + s]; __syncthreads(); }
  float rs = rsqrtf(red[0] / (float)Dc + 1e-6f);
#pragma unroll
  for (int j = 0; j < 8; ++j) {
    float y = w[base + j] * v[j] * rs;
    if (ob) ob[(size_t)row * Dc + base + j] = f2bf(y);
    if (of) of[(size_t)row * Dc + base + j] = y;
  }
}

__global__ __launch_bounds__(256) void double_rms_k(const float* __restrict__ x,
    const float* __restrict__ w2, const float* __restrict__ wm,
    float* __restrict__ of, unsigned short* __restrict__ ob) {
  __shared__ float red[256];
  int row = blockIdx.x;
  const float* xr = x + (size_t)row * Dc;
  int base = threadIdx.x * 8;
  float v[8], ss = 0.0f;
#pragma unroll
  for (int j = 0; j < 8; ++j) { v[j] = xr[base + j]; ss += v[j] * v[j]; }
  red[threadIdx.x] = ss; __syncthreads();
  for (int s = 128; s > 0; s >>= 1) { if ((int)threadIdx.x < s) red[threadIdx.x] += red[threadIdx.x + s]; __syncthreads(); }
  float rs1 = rsqrtf(red[0] / (float)Dc + 1e-6f);
  __syncthreads();
  float y[8], ss2 = 0.0f;
#pragma unroll
  for (int j = 0; j < 8; ++j) { y[j] = w2[base + j] * v[j] * rs1; ss2 += y[j] * y[j]; }
  red[threadIdx.x] = ss2; __syncthreads();
  for (int s = 128; s > 0; s >>= 1) { if ((int)threadIdx.x < s) red[threadIdx.x] += red[threadIdx.x + s]; __syncthreads(); }
  float rs2 = rsqrtf(red[0] / (float)Dc + 1e-6f);
#pragma unroll
  for (int j = 0; j < 8; ++j) {
    float z = wm[base + j] * y[j] * rs2;
    of[(size_t)row * Dc + base + j] = z;
    ob[(size_t)row * Dc + base + j] = f2bf(z);
  }
}

__global__ __launch_bounds__(256) void softmax_row(const float* __restrict__ s, unsigned short* __restrict__ o, int ncol) {
  __shared__ float red[256];
  int row = blockIdx.x;
  const float* sr = s + (size_t)row * ncol;
  int base = threadIdx.x * 4;
  float v0 = sr[base], v1 = sr[base + 1], v2 = sr[base + 2], v3 = sr[base + 3];
  float mx = fmaxf(fmaxf(v0, v1), fmaxf(v2, v3));
  red[threadIdx.x] = mx; __syncthreads();
  for (int t = 128; t > 0; t >>= 1) { if ((int)threadIdx.x < t) red[threadIdx.x] = fmaxf(red[threadIdx.x], red[threadIdx.x + t]); __syncthreads(); }
  float M = red[0]; __syncthreads();
  v0 = expf(v0 - M); v1 = expf(v1 - M); v2 = expf(v2 - M); v3 = expf(v3 - M);
  red[threadIdx.x] = v0 + v1 + v2 + v3; __syncthreads();
  for (int t = 128; t > 0; t >>= 1) { if ((int)threadIdx.x < t) red[threadIdx.x] += red[threadIdx.x + t]; __syncthreads(); }
  float inv = 1.0f / red[0];
  unsigned short* orow = o + (size_t)row * ncol;
  orow[base] = f2bf(v0 * inv); orow[base + 1] = f2bf(v1 * inv);
  orow[base + 2] = f2bf(v2 * inv); orow[base + 3] = f2bf(v3 * inv);
}

__global__ void rope_apply(float* __restrict__ p, const float* __restrict__ cb,
                           const float* __restrict__ sb, int total) {
  int idx = blockIdx.x * 256 + threadIdx.x;
  if (idx >= total) return;
  int i = idx & 31;                       // ROT/2 = 32
  int s = (idx >> 9) & (Sc - 1);          // (idx / (32*H)) % S
  float c = cb[s * 32 + i], sn = sb[s * 32 + i];
  size_t off = (size_t)(idx >> 5) * ROTc + 2 * i;
  float x0 = p[off], x1 = p[off + 1];
  p[off]     = x0 * c - x1 * sn;
  p[off + 1] = x0 * sn + x1 * c;
}

// [B,S,H,HD]+[B,S,H,ROT] (f32) -> [B,H,S,CAT] (bf16): used for qc (A of scores)
// and kc (Bt of scores, since Bt wants [n=s][k=cat] contiguous in cat)
__global__ void pack_cat(const float* __restrict__ qf, const float* __restrict__ qrf,
                         unsigned short* __restrict__ qc, int total) {
  int idx = blockIdx.x * 256 + threadIdx.x;
  if (idx >= total) return;
  int d = idx % CATc;
  int s = (idx / CATc) % Sc;
  int h = (idx / (CATc * Sc)) % Hc;
  int b = idx / (CATc * Sc * Hc);
  int t = b * Sc + s;
  float v = (d < HDc) ? qf[(size_t)t * DHc + h * HDc + d]
                      : qrf[(size_t)t * (Hc * ROTc) + h * ROTc + (d - HDc)];
  qc[idx] = f2bf(v);
}

// v [B,S,H,HD] f32 -> vt [B,H,HD,S] bf16 (Bt layout for P@V: n=d, k=s)
__global__ void pack_vt(const float* __restrict__ vf, unsigned short* __restrict__ vt, int total) {
  int idx = blockIdx.x * 256 + threadIdx.x;
  if (idx >= total) return;
  int s = idx % Sc;
  int d = (idx / Sc) % HDc;
  int h = (idx / (Sc * HDc)) % Hc;
  int b = idx / (Sc * HDc * Hc);
  vt[idx] = f2bf(vf[(size_t)(b * Sc + s) * DHc + h * HDc + d]);
}

__global__ void pack_o(const float* __restrict__ of, unsigned short* __restrict__ ob, int total) {
  int idx = blockIdx.x * 256 + threadIdx.x;
  if (idx >= total) return;
  int d2 = idx % DHc;
  int t  = idx / DHc;
  int h = d2 / HDc, d = d2 % HDc;
  int b = t / Sc, s = t % Sc;
  ob[idx] = f2bf(of[(size_t)((b * Hc + h) * Sc + s) * HDc + d]);
}

__global__ __launch_bounds__(256) void router_topk(const float* __restrict__ hn, const float* __restrict__ cent,
                                                   int* __restrict__ idx2, float* __restrict__ gate2) {
  __shared__ float red[256];
  __shared__ float affs[Ec];
  int t = blockIdx.x;
  const float* xr = hn + (size_t)t * Dc;
  int base = threadIdx.x * 8;
  float xv[8];
#pragma unroll
  for (int j = 0; j < 8; ++j) xv[j] = xr[base + j];
  for (int e = 0; e < Ec; ++e) {
    float p = 0.0f;
#pragma unroll
    for (int j = 0; j < 8; ++j) p += xv[j] * cent[(size_t)e * Dc + base + j];
    red[threadIdx.x] = p; __syncthreads();
    for (int s = 128; s > 0; s >>= 1) { if ((int)threadIdx.x < s) red[threadIdx.x] += red[threadIdx.x + s]; __syncthreads(); }
    if (threadIdx.x == 0) affs[e] = red[0];
    __syncthreads();
  }
  if (threadIdx.x == 0) {
    float m = affs[0];
    for (int e = 1; e < Ec; ++e) m = fmaxf(m, affs[e]);
    float se = 0.0f, g[Ec];
    for (int e = 0; e < Ec; ++e) { g[e] = expf(affs[e] - m); se += g[e]; }
    float inv = 1.0f / se;
    for (int e = 0; e < Ec; ++e) g[e] *= inv;
    int e0 = 0; for (int e = 1; e < Ec; ++e) if (g[e] > g[e0]) e0 = e;
    int e1 = (e0 == 0) ? 1 : 0;
    for (int e = 0; e < Ec; ++e) if (e != e0 && g[e] > g[e1]) e1 = e;
    idx2[t * 2] = e0;  gate2[t * 2] = g[e0];
    idx2[t * 2 + 1] = e1; gate2[t * 2 + 1] = g[e1];
  }
}

__global__ void build_lists(const int* __restrict__ idx2, const float* __restrict__ gate2,
                            int* __restrict__ counts, int* __restrict__ list, float* __restrict__ gates) {
  if (threadIdx.x != 0 || blockIdx.x != 0) return;
  int c[Ec];
  for (int e = 0; e < Ec; ++e) c[e] = 0;
  for (int t = 0; t < Tc; ++t)
    for (int s = 0; s < 2; ++s) {
      int e = idx2[t * 2 + s];
      int p = c[e]++;
      list[e * Tc + p] = t;
      gates[e * Tc + p] = gate2[t * 2 + s];
    }
  for (int e = 0; e < Ec; ++e) counts[e] = c[e];
}

__global__ void final_add(const float* __restrict__ xn, const float* __restrict__ hn,
                          const float* __restrict__ sh, const float* __restrict__ ro,
                          float* __restrict__ out, int n) {
  int i = blockIdx.x * 256 + threadIdx.x;
  if (i < n) out[i] = xn[i] + hn[i] + sh[i] + ro[i];
}

// ---------------- host side ----------------
static inline void gemm_launch(hipStream_t st,
    const unsigned short* A, int lda, const unsigned short* Bt, int ldbt,
    int M, int N, int K,
    const int* rowIdx, const int* cnt,
    const float* bias, const float* addsrc, float alpha, int act,
    float* Cf, unsigned short* Cb, int ldc,
    const int* stok, const float* sgate, float* sout)
{
  dim3 g(N / BN, (M + BM - 1) / BM, 1), b(256, 1, 1);
  gemm_bf16<<<g, b, 0, st>>>(A, lda, Bt, ldbt, M, N, K, rowIdx, cnt, bias, addsrc,
                             alpha, act, Cf, Cb, ldc, stok, sgate, sout);
}

extern "C" void kernel_launch(void* const* d_in, const int* in_sizes, int n_in,
                              void* d_out, int out_size, void* d_ws, size_t ws_size,
                              hipStream_t stream) {
  (void)in_sizes; (void)n_in; (void)out_size; (void)ws_size;
  const float* x      = (const float*)d_in[0];
  const float* fcos   = (const float*)d_in[1];
  const float* fsin   = (const float*)d_in[2];
  const float* wn1    = (const float*)d_in[3];
  const float* w_lq   = (const float*)d_in[4];
  const float* w_lkv  = (const float*)d_in[5];
  const float* w_q    = (const float*)d_in[6];
  const float* w_k    = (const float*)d_in[7];
  const float* w_v    = (const float*)d_in[8];
  const float* w_qr   = (const float*)d_in[9];
  const float* b_qr   = (const float*)d_in[10];
  const float* w_kr   = (const float*)d_in[11];
  const float* b_kr   = (const float*)d_in[12];
  const float* w_o    = (const float*)d_in[13];
  const float* b_o    = (const float*)d_in[14];
  const float* wn2    = (const float*)d_in[15];
  const float* wmoe   = (const float*)d_in[16];
  const float* cent   = (const float*)d_in[17];
  const float* ws1    = (const float*)d_in[18];
  const float* bs1    = (const float*)d_in[19];
  const float* ws2    = (const float*)d_in[20];
  const float* bs2    = (const float*)d_in[21];
  const float* wr1    = (const float*)d_in[22];
  const float* br1    = (const float*)d_in[23];
  const float* wr2    = (const float*)d_in[24];
  const float* br2    = (const float*)d_in[25];
  float* out = (float*)d_out;

  char* wsp = (char*)d_ws;
  size_t off = 0;
  auto alloc = [&](size_t bytes) -> void* {
    void* p = wsp + off;
    off = (off + bytes + 255) & ~(size_t)255;
    return p;
  };
  typedef unsigned short us;

  // bf16 weights, TRANSPOSED to [N,K]
  us* wlq_t  = (us*)alloc((size_t)Dc * LQc * 2);
  us* wlkv_t = (us*)alloc((size_t)Dc * LKVc * 2);
  us* wq_t   = (us*)alloc((size_t)LQc * DHc * 2);
  us* wqr_t  = (us*)alloc((size_t)LQc * Hc * ROTc * 2);
  us* wkr_t  = (us*)alloc((size_t)Dc * Hc * ROTc * 2);
  us* wk_t   = (us*)alloc((size_t)LKVc * DHc * 2);
  us* wv_t   = (us*)alloc((size_t)LKVc * DHc * 2);
  us* wo_t   = (us*)alloc((size_t)DHc * Dc * 2);
  us* ws1_t  = (us*)alloc((size_t)Dc * EDc * 2);
  us* ws2_t  = (us*)alloc((size_t)EDc * Dc * 2);
  us* wr1_t  = (us*)alloc((size_t)Ec * Dc * EDc * 2);
  us* wr2_t  = (us*)alloc((size_t)Ec * EDc * Dc * 2);
  // activations
  us* h_b    = (us*)alloc((size_t)Tc * Dc * 2);
  us* cq_b   = (us*)alloc((size_t)Tc * LQc * 2);
  us* ckv_b  = (us*)alloc((size_t)Tc * LKVc * 2);
  float* q_f  = (float*)alloc((size_t)Tc * DHc * 4);
  float* qr_f = (float*)alloc((size_t)Tc * Hc * ROTc * 4);
  float* k_f  = (float*)alloc((size_t)Tc * DHc * 4);
  float* kr_f = (float*)alloc((size_t)Tc * Hc * ROTc * 4);
  float* v_f  = (float*)alloc((size_t)Tc * DHc * 4);
  us* qc_b   = (us*)alloc((size_t)Bc * Hc * Sc * CATc * 2);   // A of scores
  us* kc_b   = (us*)alloc((size_t)Bc * Hc * Sc * CATc * 2);   // Bt of scores
  us* vt_b   = (us*)alloc((size_t)Bc * Hc * HDc * Sc * 2);    // Bt of P@V
  float* sc_f = (float*)alloc((size_t)Sc * Sc * 4);           // reused per (b,h)
  us* att_b  = (us*)alloc((size_t)Sc * Sc * 2);               // reused per (b,h)
  float* o_f  = (float*)alloc((size_t)Bc * Hc * Sc * HDc * 4);
  us* o_b    = (us*)alloc((size_t)Tc * DHc * 2);
  float* xn_f = (float*)alloc((size_t)Tc * Dc * 4);
  float* hn_f = (float*)alloc((size_t)Tc * Dc * 4);
  us* hn_b   = (us*)alloc((size_t)Tc * Dc * 2);
  us* smid_b = (us*)alloc((size_t)Tc * EDc * 2);
  float* sh_f = (float*)alloc((size_t)Tc * Dc * 4);
  int*   idx2  = (int*)alloc((size_t)Tc * 2 * 4);
  float* gate2 = (float*)alloc((size_t)Tc * 2 * 4);
  int*   counts = (int*)alloc(Ec * 4);
  int*   list   = (int*)alloc((size_t)Ec * Tc * 4);
  float* glist  = (float*)alloc((size_t)Ec * Tc * 4);
  us* midr_b = (us*)alloc((size_t)Ec * Tc * EDc * 2);
  float* routed = (float*)alloc((size_t)Tc * Dc * 4);

#define CVT_T(SRC, DST, KK, NN) do { int tot_ = (int)((size_t)(KK) * (NN)); \
    cvt_bf16_t<<<dim3((tot_ + 255) / 256), dim3(256), 0, stream>>>((SRC), (DST), (KK), (NN)); } while (0)

  // 1) weights -> bf16 transposed [N,K]
  CVT_T(w_lq,  wlq_t,  Dc, LQc);
  CVT_T(w_lkv, wlkv_t, Dc, LKVc);
  CVT_T(w_q,   wq_t,   LQc, DHc);
  CVT_T(w_qr,  wqr_t,  LQc, Hc * ROTc);
  CVT_T(w_kr,  wkr_t,  Dc, Hc * ROTc);
  CVT_T(w_k,   wk_t,   LKVc, DHc);
  CVT_T(w_v,   wv_t,   LKVc, DHc);
  CVT_T(w_o,   wo_t,   DHc, Dc);
  CVT_T(ws1,   ws1_t,  Dc, EDc);
  CVT_T(ws2,   ws2_t,  EDc, Dc);
  for (int e = 0; e < Ec; ++e) {
    CVT_T(wr1 + (size_t)e * Dc * EDc, wr1_t + (size_t)e * Dc * EDc, Dc, EDc);
    CVT_T(wr2 + (size_t)e * EDc * Dc, wr2_t + (size_t)e * EDc * Dc, EDc, Dc);
  }

  // 2) h = rms(x, w_norm1)  (bf16)
  rmsnorm_k<<<dim3(Tc), dim3(256), 0, stream>>>(x, wn1, h_b, nullptr);

  // 3) latent projections + QKV projections (all WMMA bf16)
  gemm_launch(stream, h_b, Dc, wlq_t, Dc, Tc, LQc, Dc, nullptr, nullptr, nullptr, nullptr,
              1.0f, 0, nullptr, cq_b, LQc, nullptr, nullptr, nullptr);
  gemm_launch(stream, h_b, Dc, wlkv_t, Dc, Tc, LKVc, Dc, nullptr, nullptr, nullptr, nullptr,
              1.0f, 0, nullptr, ckv_b, LKVc, nullptr, nullptr, nullptr);
  gemm_launch(stream, cq_b, LQc, wq_t, LQc, Tc, DHc, LQc, nullptr, nullptr, nullptr, nullptr,
              1.0f, 0, q_f, nullptr, DHc, nullptr, nullptr, nullptr);
  gemm_launch(stream, cq_b, LQc, wqr_t, LQc, Tc, Hc * ROTc, LQc, nullptr, nullptr, b_qr, nullptr,
              1.0f, 0, qr_f, nullptr, Hc * ROTc, nullptr, nullptr, nullptr);
  gemm_launch(stream, ckv_b, LKVc, wk_t, LKVc, Tc, DHc, LKVc, nullptr, nullptr, nullptr, nullptr,
              1.0f, 0, k_f, nullptr, DHc, nullptr, nullptr, nullptr);
  gemm_launch(stream, h_b, Dc, wkr_t, Dc, Tc, Hc * ROTc, Dc, nullptr, nullptr, b_kr, nullptr,
              1.0f, 0, kr_f, nullptr, Hc * ROTc, nullptr, nullptr, nullptr);
  gemm_launch(stream, ckv_b, LKVc, wv_t, LKVc, Tc, DHc, LKVc, nullptr, nullptr, nullptr, nullptr,
              1.0f, 0, v_f, nullptr, DHc, nullptr, nullptr, nullptr);

  // 4) RoPE on qr, kr
  {
    int total = Bc * Sc * Hc * (ROTc / 2);
    rope_apply<<<dim3((total + 255) / 256), dim3(256), 0, stream>>>(qr_f, fcos, fsin, total);
    rope_apply<<<dim3((total + 255) / 256), dim3(256), 0, stream>>>(kr_f, fcos, fsin, total);
  }

  // 5) pack attention operands (qc/kc in [B,H,S,CAT], v in [B,H,HD,S])
  {
    int tq = Bc * Hc * Sc * CATc;
    pack_cat<<<dim3((tq + 255) / 256), dim3(256), 0, stream>>>(q_f, qr_f, qc_b, tq);
    pack_cat<<<dim3((tq + 255) / 256), dim3(256), 0, stream>>>(k_f, kr_f, kc_b, tq);
    int tv = Bc * Hc * HDc * Sc;
    pack_vt<<<dim3((tv + 255) / 256), dim3(256), 0, stream>>>(v_f, vt_b, tv);
  }

  // 6) attention per (b,h): scores -> softmax -> P @ V
  const float iscale = 0.07216878364870322f; // 1/sqrt(192)
  for (int bh = 0; bh < Bc * Hc; ++bh) {
    const us* qcp = qc_b + (size_t)bh * Sc * CATc;
    const us* kcp = kc_b + (size_t)bh * Sc * CATc;
    const us* vtp = vt_b + (size_t)bh * HDc * Sc;
    float* ofp = o_f + (size_t)bh * Sc * HDc;
    gemm_launch(stream, qcp, CATc, kcp, CATc, Sc, Sc, CATc, nullptr, nullptr, nullptr, nullptr,
                iscale, 0, sc_f, nullptr, Sc, nullptr, nullptr, nullptr);
    softmax_row<<<dim3(Sc), dim3(256), 0, stream>>>(sc_f, att_b, Sc);
    gemm_launch(stream, att_b, Sc, vtp, Sc, Sc, HDc, Sc, nullptr, nullptr, nullptr, nullptr,
                1.0f, 0, ofp, nullptr, HDc, nullptr, nullptr, nullptr);
  }

  // 7) output projection + residual: xnew = x + o @ w_o + b_o
  {
    int to = Tc * DHc;
    pack_o<<<dim3((to + 255) / 256), dim3(256), 0, stream>>>(o_f, o_b, to);
  }
  gemm_launch(stream, o_b, DHc, wo_t, DHc, Tc, Dc, DHc, nullptr, nullptr, b_o, x,
              1.0f, 0, xn_f, nullptr, Dc, nullptr, nullptr, nullptr);

  // 8) hn = rms(rms(xnew, w_norm2), w_moe_norm)
  double_rms_k<<<dim3(Tc), dim3(256), 0, stream>>>(xn_f, wn2, wmoe, hn_f, hn_b);

  // 9) shared MLP: gelu((hn@ws1+bs1)@ws2+bs2)
  gemm_launch(stream, hn_b, Dc, ws1_t, Dc, Tc, EDc, Dc, nullptr, nullptr, bs1, nullptr,
              1.0f, 0, nullptr, smid_b, EDc, nullptr, nullptr, nullptr);
  gemm_launch(stream, smid_b, EDc, ws2_t, EDc, Tc, Dc, EDc, nullptr, nullptr, bs2, nullptr,
              1.0f, 1, sh_f, nullptr, Dc, nullptr, nullptr, nullptr);

  // 10) router: softmax affinities -> top-2, deterministic expert lists
  router_topk<<<dim3(Tc), dim3(256), 0, stream>>>(hn_f, cent, idx2, gate2);
  build_lists<<<dim3(1), dim3(1), 0, stream>>>(idx2, gate2, counts, list, glist);
  (void)hipMemsetAsync(routed, 0, (size_t)Tc * Dc * 4, stream);

  // 11) expert MLPs on gathered tokens only (top-2 per token)
  for (int e = 0; e < Ec; ++e) {
    gemm_launch(stream, hn_b, Dc, wr1_t + (size_t)e * Dc * EDc, Dc,
                Tc, EDc, Dc, list + e * Tc, counts + e, br1 + (size_t)e * EDc, nullptr,
                1.0f, 0, nullptr, midr_b + (size_t)e * Tc * EDc, EDc, nullptr, nullptr, nullptr);
  }
  for (int e = 0; e < Ec; ++e) {
    gemm_launch(stream, midr_b + (size_t)e * Tc * EDc, EDc, wr2_t + (size_t)e * EDc * Dc, EDc,
                Tc, Dc, EDc, nullptr, counts + e, br2 + (size_t)e * Dc, nullptr,
                1.0f, 1, nullptr, nullptr, Dc, list + e * Tc, glist + e * Tc, routed);
  }

  // 12) final: out = xnew + hn + shared + routed
  {
    int n = Tc * Dc;
    final_add<<<dim3((n + 255) / 256), dim3(256), 0, stream>>>(xn_f, hn_f, sh_f, routed, out, n);
  }
#undef CVT_T
}
